// PixelCNN_53025666237189
// MI455X (gfx1250) — matrix operand order repaired and checked by reading, not compiled, not run on hardware
//
// PixelCNN++ forward for MI455X (gfx1250), compile-only target.
//
// Every conv / nin / deconv is an implicit GEMM D[M][N] = A[M][K] x W[K][N]
// on v_wmma_f32_16x16x32_f16 (f16 inputs, f32 accumulate).
//   - activations stored f16; concat_elu / elu fused into the A stager
//     (2F=320-channel tensors never materialized)
//   - block-cooperative LDS staging of the activated A chunk: gather + ELU
//     done once per block (not once per wave); waves read fragments back as
//     two ds_load_b128 each; double-buffered slots -> 1 barrier per chunk
//   - weights pre-converted once per launch to f16, TRANSPOSED to [Npad][Kpad]
//     so each lane's B fragment is two aligned 16B loads (no gathers, no masks)
//   - K loop = (kh,kw) tap loops + 32-channel chunks: no integer division on
//     the hot path (all hot channel counts are multiples of 32)
//   - down_shift / right_shift fused as +1 pad and zeroed row0/col0
//   - stride-2 down-convs and stride-2 transposed convs via address-mode switch
//
// d_in layout assumption (documented, unverifiable at runtime):
//   [0]=row_input f32, [1]=pixel_input f32,
//   [2..413] = params leaves in jax pytree order (dict keys sorted):
//     deconv, down, down_conv, out, u_init, ul_ds, ul_rs, up
//     (each {'b','w'} leaf pair emits b then w),
//   [414]=row_start (ignored; reference uses 0).

#include <hip/hip_runtime.h>

typedef _Float16 h16;
typedef __attribute__((ext_vector_type(16))) _Float16 v16h;
typedef __attribute__((ext_vector_type(8)))  _Float16 v8h;
typedef __attribute__((ext_vector_type(8)))  float    v8f;

#define ACT_RAW  0
#define ACT_ELU  1
#define ACT_CELU 2

#define LSTRIDE 40   // LDS row stride in h16 (32 data + 8 pad; bank-conflict-free)

struct ConvDesc {
    int M;            // B * Hout * Wout
    int N;            // output channels
    int Hin, Win, Cx; // input tensor dims (stored channels; CELU doubles K)
    int Hout, Wout;
    int KH, KW;
    int padT, padL;   // transposed mode: padL is the crop offset (kw-1)/2
    int stride;       // 1 or 2 (forward conv only)
    int transposed;   // stride-2 conv_transpose gather mode
    int act;          // ACT_RAW / ACT_ELU / ACT_CELU applied to A on load
    int zr0, zc0;     // zero output contribution at row 0 / col 0 (shift fusion)
    int accum;        // 1: Y += conv + bias ; 0: Y = conv + bias
    int Kpad;         // K rounded up to 32 (B-matrix stride)
};

__device__ __forceinline__ float eluf(float x) {
    return x > 0.f ? x : (__expf(x) - 1.f);
}

__device__ __forceinline__ v8h act8(v8h a, int actmode, int side) {
#pragma unroll
    for (int e = 0; e < 8; ++e) {
        float v = (float)a[e];
        if (actmode == ACT_ELU)       v = eluf(v);
        else if (actmode == ACT_CELU) v = side ? eluf(-v) : eluf(v);
        a[e] = (h16)v;
    }
    return a;
}

// ---------------------------------------------------------------------------
// Implicit-GEMM conv kernel.
// Block = 8 waves = MT M-tiles x gpw N-groups (MT=4,gpw=2 if nG even, else 8x1).
// Fast path stages the activated A chunk (MT*16 rows x 32 K, f16) in LDS once
// per block; each wave reads its 16x32 fragment as two ds_load_b128.
// A fragment (16-bit A 16x32, ISA 7.12.2):
//   lane<16: rows M=0..15, window-K = {0..7, 16..23}; lane>=16: {8..15, 24..31}
// B fragment: lane = N column; same window-K set as A: {0..7,16..23} / {8..15,24..31},
//   two aligned v8h loads from the transposed f16 weight.
// ---------------------------------------------------------------------------
template <int NT>
__global__ __launch_bounds__(256) void k_conv_wmma(
    ConvDesc d,
    const h16* __restrict__ X,
    const h16* __restrict__ Wt,     // f16 [Npad][Kpad] transposed weights
    const float* __restrict__ Bias, // f32 [N]
    h16*   __restrict__ Yh,         // f16 out [M][N] (or null)
    float* __restrict__ Yf)         // f32 out [M][N] (or null)
{
    __shared__ h16 lds[2][128 * LSTRIDE];

    const int tid  = threadIdx.x;
    const int lane = tid & 31;
    const int wid  = tid >> 5;
    const int hi   = lane >> 4;
    const int lo   = lane & 15;

    const int Cin  = (d.act == ACT_CELU) ? 2 * d.Cx : d.Cx;
    const int ntM  = (d.M + 15) >> 4;
    const int ntN  = (d.N + 15) >> 4;
    const int nG   = (ntN + NT - 1) / NT;
    const int gpw  = ((nG & 1) == 0) ? 2 : 1;   // N-groups per block
    const int MT   = 8 / gpw;                   // M-tiles per block
    const int nGB  = nG / gpw;

    const int tmb = blockIdx.x / nGB;
    const int gb  = blockIdx.x - tmb * nGB;
    const int wm  = wid / gpw;                  // wave's M-tile within block
    const int wg  = wid - wm * gpw;             // wave's N-group within block
    const int tm  = tmb * MT + wm;
    const int tn0 = (gb * gpw + wg) * NT;
    const bool active = (tm < ntM);

    v8f acc[NT];
#pragma unroll
    for (int t = 0; t < NT; ++t)
#pragma unroll
        for (int e = 0; e < 8; ++e) acc[t][e] = 0.f;

    // per-lane B column base pointers (n < Npad by construction)
    const h16* wb[NT];
#pragma unroll
    for (int t = 0; t < NT; ++t)
        wb[t] = Wt + ((tn0 + t) * 16 + lo) * d.Kpad + (hi << 3);

    auto wmma_step = [&](const v16h& afrag, int kbase) {
#pragma unroll
        for (int t = 0; t < NT; ++t) {
            const h16* wp = wb[t] + kbase;
            const v8h b0 = *(const v8h*)wp;
            const v8h b1 = *(const v8h*)(wp + 16);
            v16h bfrag;
#pragma unroll
            for (int e = 0; e < 8; ++e) { bfrag[e] = b0[e]; bfrag[e + 8] = b1[e]; }
            acc[t] = __builtin_amdgcn_wmma_f32_16x16x32_f16(
                false, afrag, false, bfrag, (short)0, acc[t], false, false);
        }
    };

    if ((Cin & 31) == 0) {
        // ================= fast path: LDS-staged A, tap loops ==============
        const int rows = MT * 16;
        const int q    = tid & 3;               // 8-channel quarter of a chunk
        // staging rows owned by this thread (1 for MT=4, 2 for MT=8)
        int srow[2], sb[2], sh[2], sw[2]; bool smv[2];
        int scnt = 0;
        for (int r = tid >> 2; r < rows; r += 64) {
            const int mm = tmb * rows + r;
            const bool v = (mm < d.M);
            int b = 0, h = 0, w = 0;
            if (v) {
                b = mm / (d.Hout * d.Wout);
                const int rr = mm - b * d.Hout * d.Wout;
                h = rr / d.Wout;
                w = rr - h * d.Wout;
            }
            srow[scnt] = r; sb[scnt] = b; sh[scnt] = h; sw[scnt] = w;
            smv[scnt] = v; ++scnt;
        }

        int par = 0;
        for (int dh = 0; dh < d.KH; ++dh) {
            for (int dw = 0; dw < d.KW; ++dw) {
                // per-staged-row base pointer for this tap
                const h16* rp[2];
                for (int s = 0; s < scnt; ++s) {
                    rp[s] = nullptr;
                    if (smv[s]) {
                        int ih, iw; bool ok;
                        if (!d.transposed) {
                            ih = sh[s] * d.stride + dh - d.padT;
                            iw = sw[s] * d.stride + dw - d.padL;
                            ok = (ih >= 0) & (ih < d.Hin) & (iw >= 0) & (iw < d.Win);
                        } else {
                            const int i2 = sh[s] - dh;
                            const int j2 = sw[s] + d.padL - dw;   // padL == crop off
                            ih = i2 >> 1; iw = j2 >> 1;
                            ok = (i2 >= 0) & (j2 >= 0) & !(i2 & 1) & !(j2 & 1) &
                                 (ih < d.Hin) & (iw < d.Win);
                        }
                        if (ok) rp[s] = X + ((sb[s] * d.Hin + ih) * d.Win + iw) * d.Cx;
                    }
                }
                const int ktap = (dh * d.KW + dw) * Cin;
                for (int cb = 0; cb < Cin; cb += 32) {
                    const int side  = (d.act == ACT_CELU && cb >= d.Cx) ? 1 : 0;
                    const int cbase = side ? (cb - d.Cx) : cb;
                    // ---- stage: 1 global v8h load + act + 1 ds_store per row
                    for (int s = 0; s < scnt; ++s) {
                        v8h av;
                        if (rp[s]) {
                            av = *(const v8h*)(rp[s] + cbase + (q << 3));
                            av = act8(av, d.act, side);
                        } else {
#pragma unroll
                            for (int e = 0; e < 8; ++e) av[e] = (h16)0.f;
                        }
                        *(v8h*)&lds[par][srow[s] * LSTRIDE + (q << 3)] = av;
                    }
                    __syncthreads();
                    // ---- consume: fragment = two ds_load_b128
                    const int abase = (wm * 16 + lo) * LSTRIDE + (hi << 3);
                    const v8h a0 = *(const v8h*)&lds[par][abase];
                    const v8h a1 = *(const v8h*)&lds[par][abase + 16];
                    v16h afrag;
#pragma unroll
                    for (int e = 0; e < 8; ++e) { afrag[e] = a0[e]; afrag[e + 8] = a1[e]; }
                    wmma_step(afrag, ktap + cb);
                    par ^= 1;   // double buffer; next stage writes other slot
                }
            }
        }
    } else {
        // ========= generic path: only the C=3 init convs (K <= 32) =========
        const int m = tm * 16 + lo;
        const bool mval = (m < d.M);
        int ab = 0, ah = 0, aw = 0;
        if (mval) {
            ab = m / (d.Hout * d.Wout);
            const int r = m - ab * d.Hout * d.Wout;
            ah = r / d.Wout;
            aw = r - ah * d.Wout;
        }
        const int Ktot = d.KH * d.KW * Cin;
        const int kwC  = d.KW * Cin;
        for (int k0 = 0; k0 < Ktot; k0 += 32) {
            v16h afrag;
#pragma unroll
            for (int j = 0; j < 16; ++j) {
                const int v4 = j >> 1;
                const int kk = (v4 < 4) ? (2 * v4 + (j & 1) + hi * 8)
                                        : (16 + 2 * (v4 - 4) + (j & 1) + hi * 8);
                const int k = k0 + kk;
                float val = 0.f;
                if (mval && k < Ktot) {
                    const int edh = k / kwC;
                    const int er  = k - edh * kwC;
                    const int edw = er / Cin;
                    int ec        = er - edw * Cin;
                    int side = 0;
                    if (d.act == ACT_CELU && ec >= d.Cx) { side = 1; ec -= d.Cx; }
                    int ih, iw; bool v;
                    if (!d.transposed) {
                        ih = ah * d.stride + edh - d.padT;
                        iw = aw * d.stride + edw - d.padL;
                        v  = (ih >= 0) & (ih < d.Hin) & (iw >= 0) & (iw < d.Win);
                    } else {
                        const int i2 = ah - edh;
                        const int j2 = aw + d.padL - edw;
                        ih = i2 >> 1; iw = j2 >> 1;
                        v  = (i2 >= 0) & (j2 >= 0) & !(i2 & 1) & !(j2 & 1) &
                             (ih < d.Hin) & (iw < d.Win);
                    }
                    if (v) {
                        val = (float)X[((ab * d.Hin + ih) * d.Win + iw) * d.Cx + ec];
                        if (d.act == ACT_ELU)       val = eluf(val);
                        else if (d.act == ACT_CELU) val = side ? eluf(-val) : eluf(val);
                    }
                }
                afrag[j] = (h16)val;
            }
            wmma_step(afrag, k0);
        }
    }

    // ---- store (C/D layout: vgpr r -> M = r + 8*hi, N = lane&15) ----------
    if (!active) return;
#pragma unroll
    for (int t = 0; t < NT; ++t) {
        const int n = (tn0 + t) * 16 + lo;
        if (n >= d.N) continue;
        const float bv = Bias ? Bias[n] : 0.f;
#pragma unroll
        for (int r = 0; r < 8; ++r) {
            const int mm = tm * 16 + r + hi * 8;
            if (mm >= d.M) continue;
            const int bb = mm / (d.Hout * d.Wout);
            const int rr = mm - bb * d.Hout * d.Wout;
            const int hh = rr / d.Wout;
            const int ww = rr - hh * d.Wout;
            float contrib = acc[t][r] + bv;
            if ((d.zr0 && hh == 0) || (d.zc0 && ww == 0)) contrib = 0.f;
            const int idx = mm * d.N + n;
            if (Yf) {
                if (d.accum) Yf[idx] += contrib; else Yf[idx] = contrib;
            } else {
                const float prev = d.accum ? (float)Yh[idx] : 0.f;
                Yh[idx] = (h16)(prev + contrib);
            }
        }
    }
}

// out = x + h * sigmoid(g), with [h|g] = c2 (2F channels), all f16 storage
__global__ __launch_bounds__(256) void k_gate(
    const h16* __restrict__ x, const h16* __restrict__ c2,
    h16* __restrict__ out, int Mrows, int Fc)
{
    const int i = blockIdx.x * blockDim.x + threadIdx.x;
    if (i >= Mrows * Fc) return;
    const int mr = i / Fc;
    const int c  = i - mr * Fc;
    const float h = (float)c2[mr * 2 * Fc + c];
    const float g = (float)c2[mr * 2 * Fc + Fc + c];
    out[i] = (h16)((float)x[i] + h * (1.f / (1.f + __expf(-g))));
}

__global__ __launch_bounds__(256) void k_f32_to_f16(
    const float* __restrict__ in, h16* __restrict__ out, int n)
{
    const int i = blockIdx.x * blockDim.x + threadIdx.x;
    if (i < n) out[i] = (h16)in[i];
}

// transpose + convert one weight: f32 [K][N] (HWIO flat) -> f16 [Npad][Kpad]
__global__ __launch_bounds__(256) void k_prep_w(
    const float* __restrict__ W, h16* __restrict__ Wt,
    int K, int N, int Kpad, int Npad)
{
    const int i = blockIdx.x * blockDim.x + threadIdx.x;
    if (i >= Npad * Kpad) return;
    const int n = i / Kpad;
    const int k = i - n * Kpad;
    const float v = (k < K && n < N) ? W[k * N + n] : 0.f;
    Wt[i] = (h16)v;
}

// ---------------------------------------------------------------------------
struct WPrep { const h16* wt; const float* b; int Kpad; };

static void conv_launch(hipStream_t s, ConvDesc d, const WPrep& w, const h16* X,
                        h16* Yh, float* Yf)
{
    d.Kpad = w.Kpad;
    const int ntM = (d.M + 15) >> 4;
    const int ntN = (d.N + 15) >> 4;
    int NT;
    if ((ntN % 5) == 0) NT = 5;
    else if (ntN <= 7)  NT = 7;
    else                NT = 1;
    const int nG  = (ntN + NT - 1) / NT;
    const int gpw = ((nG & 1) == 0) ? 2 : 1;
    const int MT  = 8 / gpw;
    const int nGB = nG / gpw;
    const int blocks = ((ntM + MT - 1) / MT) * nGB;
    if (NT == 5)
        k_conv_wmma<5><<<dim3(blocks), dim3(256), 0, s>>>(d, X, w.wt, w.b, Yh, Yf);
    else if (NT == 7)
        k_conv_wmma<7><<<dim3(blocks), dim3(256), 0, s>>>(d, X, w.wt, w.b, Yh, Yf);
    else
        k_conv_wmma<1><<<dim3(blocks), dim3(256), 0, s>>>(d, X, w.wt, w.b, Yh, Yf);
}

extern "C" void kernel_launch(void* const* d_in, const int* in_sizes, int n_in,
                              void* d_out, int out_size, void* d_ws, size_t ws_size,
                              hipStream_t stream)
{
    (void)in_sizes; (void)n_in; (void)out_size; (void)ws_size;

    const int B_ = 36, F = 160, NMIX = 100;
    const int Hl[3] = {32, 16, 8};
    const int Ml[3] = {36 * 32 * 32, 36 * 16 * 16, 36 * 8 * 8};
    const int DNn[3] = {5, 6, 6};

    // ---------------- workspace bump allocator -----------------------------
    char* wsp = (char*)d_ws;
    size_t off = 0;
    auto alloc_h = [&](size_t elems) -> h16* {
        off = (off + 255) & ~(size_t)255;
        h16* p = (h16*)(wsp + off);
        off += elems * sizeof(h16);
        return p;
    };

    h16* xrow = alloc_h((size_t)Ml[0] * 3);
    h16* xpix = alloc_h((size_t)Ml[0] * 3);
    h16 *us[18], *uls[18];
    for (int i = 0; i < 18; ++i) {
        const int L = (i < 6) ? 0 : (i < 12) ? 1 : 2;
        us[i]  = alloc_h((size_t)Ml[L] * F);
        uls[i] = alloc_h((size_t)Ml[L] * F);
    }
    h16 *du[2], *dul[2];
    for (int i = 0; i < 2; ++i) {
        du[i]  = alloc_h((size_t)Ml[0] * F);
        dul[i] = alloc_h((size_t)Ml[0] * F);
    }
    h16* c1t = alloc_h((size_t)Ml[0] * F);
    h16* c2t = alloc_h((size_t)Ml[0] * 2 * F);

    // ---------------- parameter walk (see header for order) ----------------
    int pi = 2;
    auto nxt = [&]() { return (const float*)d_in[pi++]; };
    struct PW { const float* b; const float* w; };
    auto pw = [&]() { PW p; p.b = nxt(); p.w = nxt(); return p; };

    PW dec_u[2], dec_ul[2];
    for (int i = 0; i < 2; ++i) { dec_u[i] = pw(); dec_ul[i] = pw(); }
    struct RNd { PW uc1, uc2, una, ulc1, ulc2, ulna, ulnb; };
    RNd dn[17];
    for (int i = 0; i < 17; ++i) {
        dn[i].uc1 = pw(); dn[i].uc2 = pw(); dn[i].una = pw();
        dn[i].ulc1 = pw(); dn[i].ulc2 = pw(); dn[i].ulna = pw(); dn[i].ulnb = pw();
    }
    PW dc_u[2], dc_ul[2];
    for (int i = 0; i < 2; ++i) { dc_u[i] = pw(); dc_ul[i] = pw(); }
    PW pout   = pw();
    PW u_init = pw();
    PW ul_ds  = pw();
    PW ul_rs  = pw();
    struct RNu { PW uc1, uc2, ulc1, ulc2, ulna; };
    RNu up[15];
    for (int i = 0; i < 15; ++i) {
        up[i].uc1 = pw(); up[i].uc2 = pw();
        up[i].ulc1 = pw(); up[i].ulc2 = pw(); up[i].ulna = pw();
    }

    // ---------------- weight prep: f32 [K][N] -> f16 [Npad][Kpad] ----------
    auto prep = [&](const PW& p, int K, int N) -> WPrep {
        const int Kpad = (K + 31) & ~31;
        const int Npad = (N + 15) & ~15;
        h16* wt = alloc_h((size_t)Kpad * Npad);
        const int tot = Kpad * Npad;
        k_prep_w<<<dim3((tot + 255) / 256), dim3(256), 0, stream>>>(
            p.w, wt, K, N, Kpad, Npad);
        return WPrep{wt, p.b, Kpad};
    };

    WPrep wdec_u[2], wdec_ul[2], wdc_u[2], wdc_ul[2];
    for (int i = 0; i < 2; ++i) {
        wdec_u[i]  = prep(dec_u[i],  2 * 3 * F, F);
        wdec_ul[i] = prep(dec_ul[i], 2 * 2 * F, F);
        wdc_u[i]   = prep(dc_u[i],   2 * 3 * F, F);
        wdc_ul[i]  = prep(dc_ul[i],  2 * 2 * F, F);
    }
    struct RNdW { WPrep uc1, uc2, una, ulc1, ulc2, ulna, ulnb; };
    RNdW dnw[17];
    for (int i = 0; i < 17; ++i) {
        dnw[i].uc1  = prep(dn[i].uc1,  2 * 3 * 2 * F, F);
        dnw[i].uc2  = prep(dn[i].uc2,  2 * 3 * 2 * F, 2 * F);
        dnw[i].una  = prep(dn[i].una,  2 * F, F);
        dnw[i].ulc1 = prep(dn[i].ulc1, 2 * 2 * 2 * F, F);
        dnw[i].ulc2 = prep(dn[i].ulc2, 2 * 2 * 2 * F, 2 * F);
        dnw[i].ulna = prep(dn[i].ulna, 2 * F, F);
        dnw[i].ulnb = prep(dn[i].ulnb, 2 * F, F);
    }
    struct RNuW { WPrep uc1, uc2, ulc1, ulc2, ulna; };
    RNuW upw[15];
    for (int i = 0; i < 15; ++i) {
        upw[i].uc1  = prep(up[i].uc1,  2 * 3 * 2 * F, F);
        upw[i].uc2  = prep(up[i].uc2,  2 * 3 * 2 * F, 2 * F);
        upw[i].ulc1 = prep(up[i].ulc1, 2 * 2 * 2 * F, F);
        upw[i].ulc2 = prep(up[i].ulc2, 2 * 2 * 2 * F, 2 * F);
        upw[i].ulna = prep(up[i].ulna, 2 * F, F);
    }
    const WPrep wout    = prep(pout,   F, NMIX);
    const WPrep wu_init = prep(u_init, 2 * 3 * 3, F);
    const WPrep wul_ds  = prep(ul_ds,  1 * 3 * 3, F);
    const WPrep wul_rs  = prep(ul_rs,  2 * 1 * 3, F);

    auto mk = [&](int Lin, int Lout, int Cx, int N, int KH, int KW, int padT,
                  int padL, int stride, int transposed, int act, int zr0, int zc0,
                  int accum) {
        ConvDesc d;
        d.Hin = Hl[Lin]; d.Win = Hl[Lin]; d.Cx = Cx;
        d.Hout = Hl[Lout]; d.Wout = Hl[Lout];
        d.M = B_ * d.Hout * d.Wout; d.N = N;
        d.KH = KH; d.KW = KW; d.padT = padT; d.padL = padL;
        d.stride = stride; d.transposed = transposed; d.act = act;
        d.zr0 = zr0; d.zc0 = zc0; d.accum = accum; d.Kpad = 0;
        return d;
    };

    // ---------------- input conversion -------------------------------------
    {
        const int n = Ml[0] * 3;
        k_f32_to_f16<<<dim3((n + 255) / 256), dim3(256), 0, stream>>>(
            (const float*)d_in[0], xrow, n);
        k_f32_to_f16<<<dim3((n + 255) / 256), dim3(256), 0, stream>>>(
            (const float*)d_in[1], xpix, n);
    }

    // ---------------- initial causal convs (shifts fused) ------------------
    conv_launch(stream, mk(0, 0, 3, F, 2, 3, 2, 1, 1, 0, ACT_RAW, 1, 0, 0),
                wu_init, xrow, us[0], nullptr);
    conv_launch(stream, mk(0, 0, 3, F, 1, 3, 1, 1, 1, 0, ACT_RAW, 1, 0, 0),
                wul_ds, xrow, uls[0], nullptr);
    conv_launch(stream, mk(0, 0, 3, F, 2, 1, 1, 1, 1, 0, ACT_RAW, 0, 1, 1),
                wul_rs, xpix, uls[0], nullptr);

    // ---------------- gated resnet helper ----------------------------------
    auto gated = [&](int L, h16* x, const WPrep& c1, const WPrep& c2, int KH, int KW,
                     const WPrep* na, const h16* a, const WPrep* nb, const h16* bb,
                     h16* out) {
        conv_launch(stream, mk(L, L, F, F, KH, KW, KH - 1, 1, 1, 0, ACT_CELU, 0, 0, 0),
                    c1, x, c1t, nullptr);
        if (na)
            conv_launch(stream, mk(L, L, F, F, 1, 1, 0, 0, 1, 0, ACT_CELU, 0, 0, 1),
                        *na, a, c1t, nullptr);
        if (nb)
            conv_launch(stream, mk(L, L, F, F, 1, 1, 0, 0, 1, 0, ACT_CELU, 0, 0, 1),
                        *nb, bb, c1t, nullptr);
        conv_launch(stream, mk(L, L, F, 2 * F, KH, KW, KH - 1, 1, 1, 0, ACT_CELU, 0, 0, 0),
                    c2, c1t, c2t, nullptr);
        const int tot = Ml[L] * F;
        k_gate<<<dim3((tot + 255) / 256), dim3(256), 0, stream>>>(x, c2t, out, Ml[L], F);
    };

    // ---------------- up pass ----------------------------------------------
    int top = 0, upi = 0;
    for (int L = 0; L < 3; ++L) {
        for (int i = 0; i < 5; ++i, ++upi) {
            gated(L, us[top], upw[upi].uc1, upw[upi].uc2, 2, 3,
                  nullptr, nullptr, nullptr, nullptr, us[top + 1]);
            gated(L, uls[top], upw[upi].ulc1, upw[upi].ulc2, 2, 2,
                  &upw[upi].ulna, us[top + 1], nullptr, nullptr, uls[top + 1]);
            ++top;
        }
        if (L < 2) {
            conv_launch(stream, mk(L, L + 1, F, F, 2, 3, 1, 1, 2, 0, ACT_RAW, 0, 0, 0),
                        wdc_u[L], us[top], us[top + 1], nullptr);
            conv_launch(stream, mk(L, L + 1, F, F, 2, 2, 1, 1, 2, 0, ACT_RAW, 0, 0, 0),
                        wdc_ul[L], uls[top], uls[top + 1], nullptr);
            ++top;
        }
    }
    // top == 17

    // ---------------- down pass --------------------------------------------
    h16* ucur = us[17];
    h16* ulcur = uls[17];
    int sp = 17, pp = 0, dni = 0;
    for (int dl = 0; dl < 3; ++dl) {
        const int L = 2 - dl;
        for (int i = 0; i < DNn[dl]; ++i, ++dni) {
            --sp;
            gated(L, ucur, dnw[dni].uc1, dnw[dni].uc2, 2, 3,
                  &dnw[dni].una, us[sp], nullptr, nullptr, du[pp]);
            ucur = du[pp];
            gated(L, ulcur, dnw[dni].ulc1, dnw[dni].ulc2, 2, 2,
                  &dnw[dni].ulna, ucur, &dnw[dni].ulnb, uls[sp], dul[pp]);
            ulcur = dul[pp];
            pp ^= 1;
        }
        if (dl < 2) {
            // ds_deconv: stride-2 transposed conv; padL carries crop offset
            conv_launch(stream, mk(L, L - 1, F, F, 2, 3, 0, 1, 2, 1, ACT_RAW, 0, 0, 0),
                        wdec_u[dl], ucur, du[pp], nullptr);
            ucur = du[pp];
            conv_launch(stream, mk(L, L - 1, F, F, 2, 2, 0, 0, 2, 1, ACT_RAW, 0, 0, 0),
                        wdec_ul[dl], ulcur, dul[pp], nullptr);
            ulcur = dul[pp];
            pp ^= 1;
        }
    }

    // ---------------- output head: nin(elu(ul)) -> f32 d_out ---------------
    conv_launch(stream, mk(0, 0, F, NMIX, 1, 1, 0, 0, 1, 0, ACT_ELU, 0, 0, 0),
                wout, ulcur, nullptr, (float*)d_out);
}
